// QSRGAN_35115652612236
// MI455X (gfx1250) — matrix operand order repaired and checked by reading, not compiled, not run on hardware
//
#include <hip/hip_runtime.h>
#include <hip/hip_bf16.h>

typedef __attribute__((ext_vector_type(16))) _Float16 v16h;
typedef __attribute__((ext_vector_type(8)))  float    v8f;

#define DIM      32
#define KEEP     16
#define NGEN     4
#define QDEPTH   6
#define NQ       5
#define IN_W     256
#define OUT_H    512
#define OUT_W    512
#define AF_G_STRIDE (32 * 16)   // halfs per generator A-fragment (32 lanes x 16 halfs)

__device__ __forceinline__ float cz_sign(int i) {
  float s = 1.0f;
#pragma unroll
  for (int y = 0; y < NQ - 1; ++y) {
    int b0 = (i >> (4 - y)) & 1;
    int b1 = (i >> (3 - y)) & 1;
    s = (b0 & b1) ? -s : s;
  }
  return s;
}

// Precompute: apply the shared-weight circuit to identity columns. Thread t owns
// column t of W entirely in registers (RY mixes elements *within* a column).
// Only rows 0..15 of W are needed (the sum-normalization cancels in the
// reference). Scatter into the A-fragment order this code uses (contiguous k, same as B):
// A[m][k] = W[m][k]; lane = m + 16*((k>>4)&1);
// half h = k & 15.
__global__ void QSRGAN_build_afrag(const float* __restrict__ qp,
                                   _Float16* __restrict__ afrag) {
  const int g = blockIdx.x;   // generator
  const int t = threadIdx.x;  // column k of W (0..31)
  float st[DIM];
#pragma unroll
  for (int i = 0; i < DIM; ++i) st[i] = (i == t) ? 1.0f : 0.0f;
  const float* w = qp + g * (QDEPTH * NQ);
#pragma unroll
  for (int d = 0; d < QDEPTH; ++d) {
#pragma unroll
    for (int q = 0; q < NQ; ++q) {
      float th = 0.5f * w[d * NQ + q];
      float c = __cosf(th);
      float s = __sinf(th);
      const int off = 1 << (4 - q);
#pragma unroll
      for (int i = 0; i < DIM; ++i) {
        if ((i & off) == 0) {
          float a0 = st[i];
          float a1 = st[i + off];
          st[i]       = c * a0 - s * a1;
          st[i + off] = s * a0 + c * a1;
        }
      }
    }
#pragma unroll
    for (int i = 0; i < DIM; ++i) st[i] *= cz_sign(i);
  }
  const int hi_a = (t >> 4) & 1;
  const int h = t & 15;
  _Float16* base = afrag + g * AF_G_STRIDE;
#pragma unroll
  for (int i = 0; i < KEEP; ++i) {
    base[(i + (hi_a << 4)) * 16 + h] = (_Float16)st[i];
  }
}

// xor-16 exchange within the wave: group-of-32 ds_swizzle, xor=0x10, and=0x1f.
__device__ __forceinline__ float swz_xor16(float v) {
  return __int_as_float(__builtin_amdgcn_ds_swizzle(__float_as_int(v), 0x401F));
}

// Main kernel: one wave = 16 consecutive output pixels, all 4 generators.
// B fragment = per-pixel encoded statevector (tensor product of half-angle
// cos/sin from bilinearly upsampled angles), B layout lane: n=lane&15,
// K = (lane>>4)*16 + h. A fragment = constant W rows from d_ws. One
// v_wmma_f32_16x16x32_f16 per generator; D gives each pixel's outputs to
// 2 lanes -> in-register max tree + single ds_swizzle + b128 stores.
__global__ __launch_bounds__(256) void QSRGAN_qgen_wmma(
    const float* __restrict__ x,
    const _Float16* __restrict__ afrag,
    float* __restrict__ out) {
  const int wave = (int)((blockIdx.x * blockDim.x + threadIdx.x) >> 5);
  const int lane = threadIdx.x & 31;
  const int n  = lane & 15;   // pixel column (N) owned by this lane
  const int hi = lane >> 4;   // K bank (B) / row bank (D)

  const int tile = wave;                       // 16 consecutive pixels per tile
  const int px = ((tile & 31) << 4) + n;       // OUT_W/16 = 32 tiles per row
  const int py = tile >> 5;

  // Bilinear upsample (half-pixel centers, edge clamp): src = out*0.5 - 0.25
  float fx = (float)px * 0.5f - 0.25f;
  float fy = (float)py * 0.5f - 0.25f;
  float fx0 = floorf(fx), fy0 = floorf(fy);
  float wx = fx - fx0, wy = fy - fy0;
  int ifx = (int)fx0, ify = (int)fy0;
  int ix0 = min(IN_W - 1, max(0, ifx));
  int ix1 = min(IN_W - 1, max(0, ifx + 1));
  int iy0 = min(IN_W - 1, max(0, ify));
  int iy1 = min(IN_W - 1, max(0, ify + 1));

  const float* r00 = x + (iy0 * IN_W + ix0) * NQ;
  const float* r01 = x + (iy0 * IN_W + ix1) * NQ;
  const float* r10 = x + (iy1 * IN_W + ix0) * NQ;
  const float* r11 = x + (iy1 * IN_W + ix1) * NQ;

  float c5[NQ], s5[NQ];
#pragma unroll
  for (int c = 0; c < NQ; ++c) {
    float a = (1.0f - wy) * ((1.0f - wx) * r00[c] + wx * r01[c])
            +          wy * ((1.0f - wx) * r10[c] + wx * r11[c]);
    float ha = 0.5f * a;
    c5[c] = __cosf(ha);
    s5[c] = __sinf(ha);
  }

  // B fragment: K = hi*16 + h, so K bit4 = hi (hoisted), K bits3..0 = h.
  // state0[K] = prod over qubits of (bit set ? sin : cos); bit4<->q0 ... bit0<->q4.
  const float base4 = hi ? s5[0] : c5[0];
  v16h bstate;
#pragma unroll
  for (int h = 0; h < 16; ++h) {
    float f = base4
            * ((h & 8) ? s5[1] : c5[1])
            * ((h & 4) ? s5[2] : c5[2])
            * ((h & 2) ? s5[3] : c5[3])
            * ((h & 1) ? s5[4] : c5[4]);
    bstate[h] = (_Float16)f;
  }

  const v16h* ap = (const v16h*)afrag;  // one v16h per lane per generator

  v8f acc[NGEN];
#pragma unroll
  for (int g = 0; g < NGEN; ++g) {
    v16h aW = ap[g * 32 + lane];
    v8f c = {};
    acc[g] = __builtin_amdgcn_wmma_f32_16x16x32_f16(
        /*neg_a=*/false, aW, /*neg_b=*/false, bstate,
        /*c_mod=*/(short)0, c, /*reuse_a=*/false, /*reuse_b=*/false);
  }

  // Epilogue: out = square / max(16 kept squares). D layout: lane holds pixel
  // n = lane&15, output elements M = r + 8*hi. Other 8 elements of the same
  // pixel live in lane^16 -> one xor-16 swizzle completes the max.
  const size_t pix = (size_t)tile * 16 + n;
#pragma unroll
  for (int g = 0; g < NGEN; ++g) {
    float q[8];
#pragma unroll
    for (int r = 0; r < 8; ++r) q[r] = acc[g][r] * acc[g][r];
    float mx = fmaxf(fmaxf(fmaxf(q[0], q[1]), fmaxf(q[2], q[3])),
                     fmaxf(fmaxf(q[4], q[5]), fmaxf(q[6], q[7])));
    mx = fmaxf(mx, swz_xor16(mx));
    float inv = __builtin_amdgcn_rcpf(mx);
    float4 lo = make_float4(q[0] * inv, q[1] * inv, q[2] * inv, q[3] * inv);
    float4 hi4 = make_float4(q[4] * inv, q[5] * inv, q[6] * inv, q[7] * inv);
    float* op = out + ((size_t)g * (OUT_H * OUT_W) + pix) * KEEP + (hi << 3);
    *(float4*)(op + 0) = lo;
    *(float4*)(op + 4) = hi4;
  }
}

extern "C" void kernel_launch(void* const* d_in, const int* in_sizes, int n_in,
                              void* d_out, int out_size, void* d_ws, size_t ws_size,
                              hipStream_t stream) {
  (void)in_sizes; (void)n_in; (void)out_size; (void)ws_size;
  const float* x  = (const float*)d_in[0];   // [256,256,5] f32
  const float* qp = (const float*)d_in[1];   // [4,30] f32
  float* out = (float*)d_out;                // [4,512,512,16] f32
  _Float16* afrag = (_Float16*)d_ws;         // 4 KB of pre-swizzled A fragments

  QSRGAN_build_afrag<<<NGEN, 32, 0, stream>>>(qp, afrag);

  const int waves = (OUT_H * OUT_W) / 16;    // 16384 tiles, one wave each
  const int block = 256;                     // 8 waves per block
  QSRGAN_qgen_wmma<<<(waves * 32) / block, block, 0, stream>>>(x, afrag, out);
}
